// MoELayer_6777458393862
// MI455X (gfx1250) — matrix operand order repaired and checked by reading, not compiled, not run on hardware
//
#include <hip/hip_runtime.h>
#include <hip/hip_bf16.h>
#include <math.h>

// ---------------- problem constants (match reference) ----------------
#define D_MODEL   1024
#define D_FF      4096
#define N_EXPERTS 8
#define N_TOKENS  8192      // B*T = 4*2048

#define MTILE   16          // tokens per workgroup
#define FCHUNK  128         // F columns per pipeline chunk
#define XP      (D_MODEL + 8)   // padded LDS pitch (bf16) -> conflict-free A loads
#define HP      (FCHUNK + 8)
#define TDM_ROWS 4          // rows staged per tensor_load_to_lds

typedef __attribute__((ext_vector_type(16))) __bf16 bf16x16;
typedef __attribute__((ext_vector_type(8)))  __bf16 bf16x8;
typedef __attribute__((ext_vector_type(8)))  float  f32x8;
typedef __attribute__((ext_vector_type(4)))  unsigned int u32x4;
typedef __attribute__((ext_vector_type(8)))  int    i32x8;
typedef __attribute__((ext_vector_type(4)))  int    i32x4;

// ---------------- WMMA helpers ----------------
// reuse_a must be an immediate -> template parameter.
template <bool REUSE_A>
__device__ __forceinline__ f32x8 wmma_bf16(bf16x16 a, bf16x16 b, f32x8 c) {
    return __builtin_amdgcn_wmma_f32_16x16x32_bf16(
        false, a, false, b, (short)0, c, REUSE_A, false);
}

// A fragment (16x32 bf16) from row-major LDS tile, pitch in elems.
__device__ __forceinline__ bf16x16 load_a_lds(const __bf16* buf, int pitch,
                                              int kk, int lane) {
    const int m    = lane & 15;
    const int half = lane >> 4;
    const __bf16* p = buf + m * pitch + kk + half * 16;
    bf16x8 lo = *(const bf16x8*)(p);        // ds_load_b128
    bf16x8 hi = *(const bf16x8*)(p + 8);
    bf16x16 r;
#pragma unroll
    for (int i = 0; i < 8; ++i) { r[i] = lo[i]; r[i + 8] = hi[i]; }
    return r;
}

// B fragment (32x16) from row-major f32 weights W[r][c], row stride ld.
// Lanes of each 16-lane half read 16 consecutive columns -> coalesced.
__device__ __forceinline__ bf16x16 load_b_glb(const float* __restrict__ base,
                                              int r0, int ld, int c0, int lane) {
    const int n    = lane & 15;
    const int half = lane >> 4;
    const float* p = base + (size_t)(r0 + half * 16) * ld + c0 + n;
    bf16x16 r;
#pragma unroll
    for (int j = 0; j < 16; ++j) r[j] = (__bf16)p[(size_t)j * ld];
    return r;
}

// ---------------- Tensor Data Mover: 2-D tile, 4-byte elements ----------------
// Builds a D# per CDNA5 ISA and issues TENSOR_LOAD_TO_LDS.
// group0: count=1 | lds_addr | global_addr | type=2. group1: data_size=4B,
// tensor_dim0/1, tile_dim0/1, tensor_dim0_stride. groups 2/3 zero (2-D tile).
__device__ __forceinline__ void tdm_load_2d_f32(unsigned lds_byte_off,
                                                const float* gptr,
                                                unsigned rows, unsigned cols,
                                                unsigned row_stride_elems) {
    unsigned long long ga = (unsigned long long)(uintptr_t)gptr;
    u32x4 g0;
    g0[0] = 1u;                                   // count=1 (valid user D#)
    g0[1] = lds_byte_off;                         // lds_addr (bytes)
    g0[2] = (unsigned)(ga & 0xFFFFFFFFull);       // global_addr[31:0]
    g0[3] = (unsigned)((ga >> 32) & 0x1FFFFFFull) // global_addr[56:32]
          | (2u << 30);                           // type=2 ("image")

    unsigned long long td0 = cols;                // tensor_dim0 (elems)
    unsigned long long td1 = rows;                // tensor_dim1
    unsigned long long t0  = cols;                // tile_dim0
    unsigned long long t1  = rows;                // tile_dim1
    unsigned long long s0  = row_stride_elems;    // tensor_dim0_stride
    unsigned long long q0 = (2ull << 16)                     // data_size = 4B
                          | ((td0 & 0xFFFFull) << 48);
    unsigned long long q1 = ((td0 >> 16) & 0xFFFFull)
                          | ((td1 & 0xFFFFFFFFull) << 16)
                          | ((t0  & 0xFFFFull)    << 48);
    unsigned long long q2 = (t1 & 0xFFFFull)                 // tile_dim2 = 0
                          | ((s0 & 0xFFFFFFFFull) << 32);
    unsigned long long q3 = (s0 >> 32) & 0xFFFFull;          // dim1_stride = 0
    i32x8 g1;
    g1[0] = (int)(q0 & 0xFFFFFFFFull); g1[1] = (int)(q0 >> 32);
    g1[2] = (int)(q1 & 0xFFFFFFFFull); g1[3] = (int)(q1 >> 32);
    g1[4] = (int)(q2 & 0xFFFFFFFFull); g1[5] = (int)(q2 >> 32);
    g1[6] = (int)(q3 & 0xFFFFFFFFull); g1[7] = (int)(q3 >> 32);
    i32x4 z4 = {0, 0, 0, 0};
#if defined(__clang_major__) && (__clang_major__ >= 23)
    i32x8 z8 = {0, 0, 0, 0, 0, 0, 0, 0};
    __builtin_amdgcn_tensor_load_to_lds(g0, g1, z4, z4, z8, 0);
#else
    __builtin_amdgcn_tensor_load_to_lds(g0, g1, z4, z4, 0);
#endif
}

// ---------------- kernel 0: zero the accumulation buffer ----------------
__global__ void zero_kernel(float* __restrict__ out, int n) {
    int i = blockIdx.x * blockDim.x + threadIdx.x;
    if (i < n) out[i] = 0.0f;
}

// ---------------- kernel 1: router (softmax + top-2 combine weights) ----------------
__global__ __launch_bounds__(256) void router_kernel(
    const float* __restrict__ x, const float* __restrict__ rw,
    float* __restrict__ cw) {
    const int lane  = threadIdx.x & 31;
    const int wave  = threadIdx.x >> 5;
    const int token = blockIdx.x * 8 + wave;
    if (token >= N_TOKENS) return;

    const float* xr = x + (size_t)token * D_MODEL;
    float acc[N_EXPERTS];
#pragma unroll
    for (int e = 0; e < N_EXPERTS; ++e) acc[e] = 0.0f;

    for (int d = lane; d < D_MODEL; d += 32) {
        float xv = xr[d];
#pragma unroll
        for (int e = 0; e < N_EXPERTS; ++e)
            acc[e] += xv * rw[e * D_MODEL + d];
    }
#pragma unroll
    for (int e = 0; e < N_EXPERTS; ++e)
        for (int off = 16; off > 0; off >>= 1)
            acc[e] += __shfl_xor(acc[e], off, 32);

    if (lane == 0) {
        float mx = acc[0];
#pragma unroll
        for (int e = 1; e < N_EXPERTS; ++e) mx = fmaxf(mx, acc[e]);
        float p[N_EXPERTS];
#pragma unroll
        for (int e = 0; e < N_EXPERTS; ++e) p[e] = __expf(acc[e] - mx);
        int i0 = 0;
#pragma unroll
        for (int e = 1; e < N_EXPERTS; ++e) if (p[e] > p[i0]) i0 = e;
        int i1 = (i0 == 0) ? 1 : 0;
#pragma unroll
        for (int e = 0; e < N_EXPERTS; ++e)
            if (e != i0 && p[e] > p[i1]) i1 = e;
        float inv = __builtin_amdgcn_rcpf(p[i0] + p[i1]);
#pragma unroll
        for (int e = 0; e < N_EXPERTS; ++e)
            cw[(size_t)token * N_EXPERTS + e] =
                (e == i0) ? p[i0] * inv : ((e == i1) ? p[i1] * inv : 0.0f);
    }
}

// ---------------- kernel 2: fused SwiGLU expert FFN ----------------
__global__ __launch_bounds__(256) void moe_expert_kernel(
    const float* __restrict__ x,
    const float* __restrict__ w_gate, const float* __restrict__ w_up,
    const float* __restrict__ w_down, const float* __restrict__ cw,
    float* __restrict__ out) {
    __shared__ float  x_stage[TDM_ROWS * D_MODEL];  // 16 KB TDM landing pad
    __shared__ __bf16 x_lds[MTILE * XP];            // ~33 KB
    __shared__ __bf16 h_lds[MTILE * HP];            // ~4.3 KB
    __shared__ float  cw_lds[MTILE];

    const int tile0 = blockIdx.x * MTILE;
    const int e     = blockIdx.y;
    const int tid   = threadIdx.x;
    const int lane  = tid & 31;
    const int wave  = tid >> 5;

    const float* wg = w_gate + (size_t)e * D_MODEL * D_FF;
    const float* wu = w_up   + (size_t)e * D_MODEL * D_FF;
    const float* wd = w_down + (size_t)e * D_FF * D_MODEL;

    // ---- stage token tile via Tensor Data Mover, convert f32 -> bf16 ----
    for (int c = 0; c < MTILE / TDM_ROWS; ++c) {
        if (wave == 0) {
            tdm_load_2d_f32((unsigned)(uintptr_t)(void*)x_stage,
                            x + (size_t)(tile0 + c * TDM_ROWS) * D_MODEL,
                            TDM_ROWS, D_MODEL, D_MODEL);
            __builtin_amdgcn_s_wait_tensorcnt(0);
        }
        __syncthreads();                       // TDM data visible to all waves
        for (int i = tid; i < TDM_ROWS * D_MODEL; i += 256) {
            int r = i >> 10, cc = i & (D_MODEL - 1);
            x_lds[(c * TDM_ROWS + r) * XP + cc] = (__bf16)x_stage[i];
        }
        __syncthreads();                       // before landing pad is reused
    }
    if (tid < MTILE)
        cw_lds[tid] = cw[(size_t)(tile0 + tid) * N_EXPERTS + e];
    __syncthreads();

    // Persistent down-proj accumulators: wave owns out cols [wave*128, +128).
    f32x8 acc[8];
#pragma unroll
    for (int t = 0; t < 8; ++t)
#pragma unroll
        for (int j = 0; j < 8; ++j) acc[t][j] = 0.0f;

    for (int fc = 0; fc < D_FF; fc += FCHUNK) {
        if (fc + FCHUNK < D_FF) {
            __builtin_prefetch(wg + (size_t)(fc + FCHUNK) + (size_t)wave * 16, 0, 1);
            __builtin_prefetch(wd + (size_t)(fc + FCHUNK) * D_MODEL + wave * 128, 0, 1);
        }

        // ---- phase 1: h[:, fc+wave*16 .. +16) = silu(x@Wg) * (x@Wu) ----
        const int f0 = fc + wave * 16;
        f32x8 gacc, uacc;
#pragma unroll
        for (int j = 0; j < 8; ++j) { gacc[j] = 0.0f; uacc[j] = 0.0f; }

        for (int kk = 0; kk < D_MODEL; kk += 32) {
            bf16x16 a  = load_a_lds(x_lds, XP, kk, lane);
            bf16x16 bg = load_b_glb(wg, kk, D_FF, f0, lane);
            bf16x16 bu = load_b_glb(wu, kk, D_FF, f0, lane);
            gacc = wmma_bf16<false>(a, bg, gacc);
            uacc = wmma_bf16<true>(a, bu, uacc);   // same A as previous WMMA
        }
        // SwiGLU: g * rcp(1 + exp(-g)) * u  (v_exp_f32 + v_rcp_f32, no IEEE div)
#pragma unroll
        for (int j = 0; j < 8; ++j) {
            float g = gacc[j];
            float s = __builtin_amdgcn_rcpf(1.0f + __expf(-g));
            float h = g * s * uacc[j];
            int m = j + ((lane >> 4) << 3);
            h_lds[m * HP + wave * 16 + (lane & 15)] = (__bf16)h;
        }
        __syncthreads();

        // ---- phase 2: acc += h_chunk @ Wd[fc:fc+128, wave*128 : +128) ----
        for (int ks = 0; ks < FCHUNK; ks += 32) {
            bf16x16 a = load_a_lds(h_lds, HP, ks, lane);
            // ct = 0: fresh A operand
            {
                bf16x16 b = load_b_glb(wd, fc + ks, D_MODEL, wave * 128, lane);
                acc[0] = wmma_bf16<false>(a, b, acc[0]);
            }
            // ct = 1..7: identical instruction, same A -> reuse hint legal
#pragma unroll
            for (int ct = 1; ct < 8; ++ct) {
                bf16x16 b = load_b_glb(wd, fc + ks, D_MODEL,
                                       wave * 128 + ct * 16, lane);
                acc[ct] = wmma_bf16<true>(a, b, acc[ct]);
            }
        }
        __syncthreads();   // before h_lds is overwritten next chunk
    }

    // ---- epilogue: routing-weight row scale + cross-expert accumulation ----
#pragma unroll
    for (int ct = 0; ct < 8; ++ct) {
#pragma unroll
        for (int j = 0; j < 8; ++j) {
            int   m    = j + ((lane >> 4) << 3);
            float wtok = cw_lds[m];
            int   col  = wave * 128 + ct * 16 + (lane & 15);
            atomicAdd(out + (size_t)(tile0 + m) * D_MODEL + col,
                      wtok * acc[ct][j]);
        }
    }
}

// ---------------- host-side launch ----------------
extern "C" void kernel_launch(void* const* d_in, const int* in_sizes, int n_in,
                              void* d_out, int out_size, void* d_ws, size_t ws_size,
                              hipStream_t stream) {
    (void)in_sizes; (void)n_in; (void)ws_size;
    const float* x        = (const float*)d_in[0];
    const float* router_w = (const float*)d_in[1];
    const float* w_gate   = (const float*)d_in[2];
    const float* w_up     = (const float*)d_in[3];
    const float* w_down   = (const float*)d_in[4];
    float*       out      = (float*)d_out;
    float*       cw       = (float*)d_ws;   // [N_TOKENS][N_EXPERTS]

    zero_kernel<<<(out_size + 255) / 256, 256, 0, stream>>>(out, out_size);
    router_kernel<<<N_TOKENS / 8, 256, 0, stream>>>(x, router_w, cw);
    dim3 grid(N_TOKENS / MTILE, N_EXPERTS);
    moe_expert_kernel<<<grid, 256, 0, stream>>>(x, w_gate, w_up, w_down, cw, out);
}
